// MLA_59889023975559
// MI455X (gfx1250) — compile-verified, matrix operand order checked by reading, not run on hardware
//
#include <hip/hip_runtime.h>
#include <hip/hip_bf16.h>

// ---------------------------------------------------------------------------
// MLA prefill for MI455X (gfx1250, wave32, WMMA 16x16x32 bf16)
// B=2 S=2048 DIM=2048 H=16 QR=1536 KVR=512 DN=128 DR=64 DV=128 QKD=192
// ---------------------------------------------------------------------------

#define B_    2
#define S_    2048
#define DIM_  2048
#define H_    16
#define QR_   1536
#define KVR_  512
#define DN_   128
#define DR_   64
#define DV_   128
#define QKD_  192          // DN + DR
#define MTOT_ (B_ * S_)    // 4096 rows
#define SCALE_ 0.07216878364870323f   // 192^-0.5
#define EPS_   1.1920929e-7f

typedef __attribute__((ext_vector_type(16))) __bf16 bf16x16;
typedef __attribute__((ext_vector_type(8)))  __bf16 bf16x8;
typedef __attribute__((ext_vector_type(8)))  float  f32x8;

__device__ __forceinline__ __bf16 f2bf(float f) {
    unsigned u = __builtin_bit_cast(unsigned, f);
    u = (u + 0x7FFFu + ((u >> 16) & 1u)) >> 16;      // round-to-nearest-even
    unsigned short h = (unsigned short)u;
    return __builtin_bit_cast(__bf16, h);
}

// Load one lane's 16 halves of an A/B WMMA fragment (16x32 / 32x16 bf16).
// rowk points at this lane's row, kb = (lane>=16) ? 8 : 0.
// Halves {kb..kb+7} -> elements 0..7, {16+kb..16+kb+7} -> elements 8..15.
__device__ __forceinline__ bf16x16 frag_ab(const __bf16* rowk, int kb) {
    bf16x8 lo = *reinterpret_cast<const bf16x8*>(rowk + kb);
    bf16x8 hi = *reinterpret_cast<const bf16x8*>(rowk + 16 + kb);
    bf16x16 r;
#pragma unroll
    for (int i = 0; i < 8; ++i) { r[i] = lo[i]; r[8 + i] = hi[i]; }
    return r;
}

// ---------------------------------------------------------------------------
// fp32 -> bf16 conversion
// ---------------------------------------------------------------------------
__global__ __launch_bounds__(256) void cvt_bf16_kernel(
    const float* __restrict__ in, __bf16* __restrict__ out, int n)
{
    int i = blockIdx.x * 256 + threadIdx.x;
    if (i < n) out[i] = f2bf(in[i]);
}

// ---------------------------------------------------------------------------
// C[M,N] = A[M,K](bf16,row-major) @ W[N,K]^T(bf16,row-major) + bias[N]
// block = 128 threads (4 waves).  Wave w owns a 64x64 macro-tile:
//   rows [bx*256 + w*64, +64), cols [by*64, +64)  ->  4x4 WMMA tiles.
// Per K=32 step: 4 A-frags + 4 B-frags (16x global_load_b128) -> 16 WMMAs,
// i.e. 1:1 load:wmma and 4x register reuse of every fragment.
// ---------------------------------------------------------------------------
__global__ __launch_bounds__(128) void gemm_bf16_kernel(
    const __bf16* __restrict__ A, const __bf16* __restrict__ W,
    const float* __restrict__ bias, float* __restrict__ C,
    int M, int N, int K)
{
    const int lane = threadIdx.x & 31;
    const int wave = threadIdx.x >> 5;
    const int m0   = blockIdx.x * 256 + wave * 64;
    const int n0   = blockIdx.y * 64;
    const int lr   = lane & 15;
    const int hi   = lane >> 4;
    const int kb   = hi << 3;

    const __bf16* arow[4];
    const __bf16* wrow[4];
#pragma unroll
    for (int i = 0; i < 4; ++i) {
        arow[i] = A + (size_t)(m0 + i * 16 + lr) * K;
        wrow[i] = W + (size_t)(n0 + i * 16 + lr) * K;
    }

    f32x8 acc[16];
#pragma unroll
    for (int t = 0; t < 16; ++t) acc[t] = (f32x8){};

    for (int kk = 0; kk < K; kk += 32) {
        __builtin_prefetch(arow[0] + kk + 256, 0, 1);
        bf16x16 af[4], bfr[4];
#pragma unroll
        for (int i = 0; i < 4; ++i) af[i]  = frag_ab(arow[i] + kk, kb);
#pragma unroll
        for (int i = 0; i < 4; ++i) bfr[i] = frag_ab(wrow[i] + kk, kb);
#pragma unroll
        for (int mi = 0; mi < 4; ++mi)
#pragma unroll
            for (int ni = 0; ni < 4; ++ni)
                acc[mi * 4 + ni] = __builtin_amdgcn_wmma_f32_16x16x32_bf16(
                    false, af[mi], false, bfr[ni], (short)0,
                    acc[mi * 4 + ni], false, false);
    }

    float bn[4];
#pragma unroll
    for (int ni = 0; ni < 4; ++ni) bn[ni] = bias[n0 + ni * 16 + lr];

#pragma unroll
    for (int mi = 0; mi < 4; ++mi)
#pragma unroll
        for (int v = 0; v < 8; ++v) {
            const int row = m0 + mi * 16 + v + 8 * hi;
#pragma unroll
            for (int ni = 0; ni < 4; ++ni)
                C[(size_t)row * N + n0 + ni * 16 + lr] =
                    acc[mi * 4 + ni][v] + bn[ni];
        }
}

// ---------------------------------------------------------------------------
// out[row, 0..width) = in[row*ld .. +width) * rsqrt(mean(x^2)+eps) * w  (bf16)
// ---------------------------------------------------------------------------
__global__ __launch_bounds__(256) void rmsnorm_bf16_kernel(
    const float* __restrict__ in, int ld, int width,
    const float* __restrict__ w, __bf16* __restrict__ out)
{
    __shared__ float red[8];
    __shared__ float rtot;
    const int row = blockIdx.x;
    const float* p = in + (size_t)row * ld;

    float ss = 0.f;
    for (int i = threadIdx.x; i < width; i += 256) { float v = p[i]; ss += v * v; }
#pragma unroll
    for (int off = 16; off > 0; off >>= 1) ss += __shfl_xor(ss, off, 32);
    if ((threadIdx.x & 31) == 0) red[threadIdx.x >> 5] = ss;
    __syncthreads();
    if (threadIdx.x == 0) {
        float t = 0.f;
#pragma unroll
        for (int i = 0; i < 8; ++i) t += red[i];
        rtot = rsqrtf(t / (float)width + EPS_);
    }
    __syncthreads();
    const float r = rtot;
    for (int i = threadIdx.x; i < width; i += 256)
        out[(size_t)row * width + i] = f2bf(p[i] * r * w[i]);
}

// ---------------------------------------------------------------------------
// q fp32 [MTOT, H*192] -> q_bf [MTOT, H, 192], rope on last 64 dims per head
// grid (MTOT, H), block 96
// ---------------------------------------------------------------------------
__global__ __launch_bounds__(96) void assemble_q_kernel(
    const float* __restrict__ q, const float* __restrict__ fc,
    const float* __restrict__ fs, __bf16* __restrict__ qb)
{
    const int r = blockIdx.x, h = blockIdx.y;
    const int s = r % S_;
    const float* src = q + (size_t)r * (H_ * QKD_) + h * QKD_;
    __bf16* dst = qb + ((size_t)r * H_ + h) * QKD_;
    const int t = threadIdx.x;
    if (t < 64) {
        dst[2 * t]     = f2bf(src[2 * t]);
        dst[2 * t + 1] = f2bf(src[2 * t + 1]);
    } else {
        const int j = t - 64;                 // 0..31
        float xr = src[DN_ + 2 * j], xi = src[DN_ + 2 * j + 1];
        float c = fc[s * (DR_ / 2) + j], sn = fs[s * (DR_ / 2) + j];
        dst[DN_ + 2 * j]     = f2bf(xr * c - xi * sn);
        dst[DN_ + 2 * j + 1] = f2bf(xr * sn + xi * c);
    }
}

// ---------------------------------------------------------------------------
// kv fp32 [MTOT, H*256] + kv_a fp32 [MTOT, 576] ->
//   k_bf [MTOT, H, 192]  (k_nope | roped k_pe broadcast)
//   v_t  bf16 [B, H, 128, S]   (transposed for PV B-fragments)
// grid (MTOT, H), block 128
// ---------------------------------------------------------------------------
__global__ __launch_bounds__(128) void assemble_kv_kernel(
    const float* __restrict__ kv, const float* __restrict__ kva,
    const float* __restrict__ fc, const float* __restrict__ fs,
    __bf16* __restrict__ kb_, __bf16* __restrict__ vt)
{
    const int r = blockIdx.x, h = blockIdx.y;
    const int b = r / S_, s = r % S_;
    const float* kvrow = kv + (size_t)r * (H_ * (DN_ + DV_)) + h * (DN_ + DV_);
    __bf16* krow = kb_ + ((size_t)r * H_ + h) * QKD_;
    const int t = threadIdx.x;            // 0..127
    krow[t] = f2bf(kvrow[t]);                                        // k_nope
    vt[((size_t)(b * H_ + h) * DV_ + t) * S_ + s] = f2bf(kvrow[DN_ + t]); // v^T
    if (t < DR_ / 2) {
        const int j = t;
        float xr = kva[(size_t)r * (KVR_ + DR_) + KVR_ + 2 * j];
        float xi = kva[(size_t)r * (KVR_ + DR_) + KVR_ + 2 * j + 1];
        float c = fc[s * (DR_ / 2) + j], sn = fs[s * (DR_ / 2) + j];
        krow[DN_ + 2 * j]     = f2bf(xr * c - xi * sn);
        krow[DN_ + 2 * j + 1] = f2bf(xr * sn + xi * c);
    }
}

// ---------------------------------------------------------------------------
// Flash attention. grid (S/64, H, B), block 128 = 4 waves.
// Wave w owns q rows [ (bx*4+w)*16, +16 ).  Causal, 32 keys per step:
//   2x (6 WMMA QK^T chunks)  + online softmax + LDS P transpose + 8 WMMA PV.
// Output directly as bf16 [MTOT, H*128] for the final wo GEMM.
// ---------------------------------------------------------------------------
__global__ __launch_bounds__(128) void mla_attn_kernel(
    const __bf16* __restrict__ qb, const __bf16* __restrict__ kbuf,
    const __bf16* __restrict__ vt, __bf16* __restrict__ ob)
{
    __shared__ float ldsP[4][16][32];
    const int lane = threadIdx.x & 31;
    const int wave = threadIdx.x >> 5;
    const int h = blockIdx.y, b = blockIdx.z;
    const int q0 = (blockIdx.x * 4 + wave) * 16;
    const int lr = lane & 15;
    const int hi = lane >> 4;
    const int kbo = hi << 3;

    // Q fragments: 6 chunks of K=32 over the 192-dim head
    bf16x16 qf[6];
    const __bf16* qrow = qb + ((size_t)(b * S_ + q0 + lr) * H_ + h) * QKD_;
#pragma unroll
    for (int c = 0; c < 6; ++c) qf[c] = frag_ab(qrow + c * 32, kbo);

    f32x8 acc[8];
    float mI[8], lI[8];
#pragma unroll
    for (int j = 0; j < 8; ++j) acc[j] = (f32x8){};
#pragma unroll
    for (int v = 0; v < 8; ++v) { mI[v] = -3.0e38f; lI[v] = 0.f; }

    const int ktmax = (q0 + 15) >> 5;
    for (int kt = 0; kt <= ktmax; ++kt) {
        const int k32 = kt << 5;
        float p0[8], p1[8];

        {   // key sub-tile 0: keys [k32, k32+16)
            const __bf16* krow =
                kbuf + ((size_t)(b * S_ + k32 + lr) * H_ + h) * QKD_;
            f32x8 c = {};
#pragma unroll
            for (int ch = 0; ch < 6; ++ch) {
                bf16x16 kf = frag_ab(krow + ch * 32, kbo);
                c = __builtin_amdgcn_wmma_f32_16x16x32_bf16(
                    false, qf[ch], false, kf, (short)0, c, false, false);
            }
            const int tcol = k32 + lr;
#pragma unroll
            for (int v = 0; v < 8; ++v) {
                const int srow = q0 + v + 8 * hi;
                p0[v] = c[v] * SCALE_ + (tcol > srow ? -1.0e9f : 0.f);
            }
        }
        const bool sub1 = (k32 + 16) <= (q0 + 15);
        if (sub1) {  // key sub-tile 1: keys [k32+16, k32+32)
            const __bf16* krow =
                kbuf + ((size_t)(b * S_ + k32 + 16 + lr) * H_ + h) * QKD_;
            f32x8 c = {};
#pragma unroll
            for (int ch = 0; ch < 6; ++ch) {
                bf16x16 kf = frag_ab(krow + ch * 32, kbo);
                c = __builtin_amdgcn_wmma_f32_16x16x32_bf16(
                    false, qf[ch], false, kf, (short)0, c, false, false);
            }
            const int tcol = k32 + 16 + lr;
#pragma unroll
            for (int v = 0; v < 8; ++v) {
                const int srow = q0 + v + 8 * hi;
                p1[v] = c[v] * SCALE_ + (tcol > srow ? -1.0e9f : 0.f);
            }
        } else {
#pragma unroll
            for (int v = 0; v < 8; ++v) p1[v] = -1.0e9f;
        }

        // online softmax: stats per row (row = v + 8*hi), reduce over 16 lanes
        float sf[8];
#pragma unroll
        for (int v = 0; v < 8; ++v) {
            float mx = fmaxf(p0[v], p1[v]);
#pragma unroll
            for (int off = 1; off < 16; off <<= 1)
                mx = fmaxf(mx, __shfl_xor(mx, off, 32));
            const float nm = fmaxf(mI[v], mx);
            sf[v] = __expf(mI[v] - nm);
            p0[v] = __expf(p0[v] - nm);
            p1[v] = __expf(p1[v] - nm);
            float rs = p0[v] + p1[v];
#pragma unroll
            for (int off = 1; off < 16; off <<= 1)
                rs += __shfl_xor(rs, off, 32);
            lI[v] = lI[v] * sf[v] + rs;
            mI[v] = nm;
        }
#pragma unroll
        for (int j = 0; j < 8; ++j)
#pragma unroll
            for (int v = 0; v < 8; ++v) acc[j][v] *= sf[v];

        // C-layout P -> A-layout bf16 fragment via per-wave LDS (16x32 fp32)
        float (*P)[32] = ldsP[wave];
#pragma unroll
        for (int v = 0; v < 8; ++v) {
            P[v + 8 * hi][lr]      = p0[v];
            P[v + 8 * hi][16 + lr] = p1[v];
        }
        bf16x16 pf;
        {
            const float* prow = P[lr];
#pragma unroll
            for (int i = 0; i < 8; ++i) {
                pf[i]     = f2bf(prow[kbo + i]);
                pf[8 + i] = f2bf(prow[16 + kbo + i]);
            }
        }

        // PV: acc[j] += P(16x32) x V(32x16) for each 16-dim chunk of DV
        const __bf16* vbase = vt + (size_t)(b * H_ + h) * DV_ * S_ + k32;
#pragma unroll
        for (int j = 0; j < 8; ++j) {
            const __bf16* vrow = vbase + (size_t)(j * 16 + lr) * S_;
            bf16x16 vf = frag_ab(vrow, kbo);
            acc[j] = __builtin_amdgcn_wmma_f32_16x16x32_bf16(
                false, pf, false, vf, (short)0, acc[j], false, false);
        }
    }

    // epilogue: normalize by row-sum, emit bf16 [MTOT, H*128]
#pragma unroll
    for (int v = 0; v < 8; ++v) {
        const float inv = 1.f / lI[v];
        const int srow = q0 + v + 8 * hi;
        __bf16* orow = ob + (size_t)(b * S_ + srow) * (H_ * DV_) + h * DV_;
#pragma unroll
        for (int j = 0; j < 8; ++j)
            orow[j * 16 + lr] = f2bf(acc[j][v] * inv);
    }
}

// ---------------------------------------------------------------------------
// host orchestration
// ---------------------------------------------------------------------------
extern "C" void kernel_launch(void* const* d_in, const int* in_sizes, int n_in,
                              void* d_out, int out_size, void* d_ws, size_t ws_size,
                              hipStream_t stream)
{
    (void)in_sizes; (void)n_in; (void)out_size; (void)ws_size;
    const float* x      = (const float*)d_in[0];
    const float* fc     = (const float*)d_in[1];
    const float* fs     = (const float*)d_in[2];
    /* mask d_in[3] handled analytically (causal) */
    const float* wqa    = (const float*)d_in[4];
    const float* wqa_b  = (const float*)d_in[5];
    const float* qnw    = (const float*)d_in[6];
    const float* wqb    = (const float*)d_in[7];
    const float* wqb_b  = (const float*)d_in[8];
    const float* wkva   = (const float*)d_in[9];
    const float* wkva_b = (const float*)d_in[10];
    const float* kvnw   = (const float*)d_in[11];
    const float* wkvb   = (const float*)d_in[12];
    const float* wkvb_b = (const float*)d_in[13];
    const float* wo     = (const float*)d_in[14];
    const float* wo_b   = (const float*)d_in[15];
    float* out = (float*)d_out;

    char* ws = (char*)d_ws;
    size_t off = 0;
    auto alloc = [&](size_t bytes) -> void* {
        void* p = ws + off;
        off = (off + bytes + 255) & ~((size_t)255);
        return p;
    };

    const size_t nX    = (size_t)MTOT_ * DIM_;
    const size_t nWqa  = (size_t)QR_ * DIM_;
    const size_t nWqb  = (size_t)H_ * QKD_ * QR_;
    const size_t nWkva = (size_t)(KVR_ + DR_) * DIM_;
    const size_t nWkvb = (size_t)H_ * (DN_ + DV_) * KVR_;
    const size_t nWo   = (size_t)DIM_ * H_ * DV_;

    __bf16* xb    = (__bf16*)alloc(nX * 2);
    __bf16* wqab  = (__bf16*)alloc(nWqa * 2);
    __bf16* wqbb  = (__bf16*)alloc(nWqb * 2);
    __bf16* wkvab = (__bf16*)alloc(nWkva * 2);
    __bf16* wkvbb = (__bf16*)alloc(nWkvb * 2);
    __bf16* wob   = (__bf16*)alloc(nWo * 2);
    float*  qa    = (float*)alloc((size_t)MTOT_ * QR_ * 4);
    __bf16* qan   = (__bf16*)alloc((size_t)MTOT_ * QR_ * 2);
    float*  qfull = (float*)alloc((size_t)MTOT_ * H_ * QKD_ * 4);
    float*  kva   = (float*)alloc((size_t)MTOT_ * (KVR_ + DR_) * 4);
    __bf16* kvn   = (__bf16*)alloc((size_t)MTOT_ * KVR_ * 2);
    float*  kvfull= (float*)alloc((size_t)MTOT_ * H_ * (DN_ + DV_) * 4);
    __bf16* qbf   = (__bf16*)alloc((size_t)MTOT_ * H_ * QKD_ * 2);
    __bf16* kbf   = (__bf16*)alloc((size_t)MTOT_ * H_ * QKD_ * 2);
    __bf16* vtb   = (__bf16*)alloc((size_t)B_ * H_ * DV_ * S_ * 2);
    __bf16* attnb = (__bf16*)alloc((size_t)MTOT_ * H_ * DV_ * 2);

    auto cvt = [&](const float* src, __bf16* dst, size_t n) {
        cvt_bf16_kernel<<<dim3((unsigned)((n + 255) / 256)), dim3(256), 0, stream>>>(
            src, dst, (int)n);
    };
    cvt(x, xb, nX);
    cvt(wqa, wqab, nWqa);
    cvt(wqb, wqbb, nWqb);
    cvt(wkva, wkvab, nWkva);
    cvt(wkvb, wkvbb, nWkvb);
    cvt(wo, wob, nWo);

    auto gemm = [&](const __bf16* A, const __bf16* W, const float* bias,
                    float* C, int M, int N, int K) {
        gemm_bf16_kernel<<<dim3(M / 256, N / 64), dim3(128), 0, stream>>>(
            A, W, bias, C, M, N, K);
    };

    // q_a = x @ wq_a^T + b  -> rmsnorm -> q = q_a_n @ wq_b^T + b
    gemm(xb, wqab, wqa_b, qa, MTOT_, QR_, DIM_);
    rmsnorm_bf16_kernel<<<dim3(MTOT_), dim3(256), 0, stream>>>(qa, QR_, QR_, qnw, qan);
    gemm(qan, wqbb, wqb_b, qfull, MTOT_, H_ * QKD_, QR_);

    // kv_a = x @ wkv_a^T + b -> rmsnorm(first 512) -> kv = kv_n @ wkv_b^T + b
    gemm(xb, wkvab, wkva_b, kva, MTOT_, KVR_ + DR_, DIM_);
    rmsnorm_bf16_kernel<<<dim3(MTOT_), dim3(256), 0, stream>>>(kva, KVR_ + DR_, KVR_, kvnw, kvn);
    gemm(kvn, wkvbb, wkvb_b, kvfull, MTOT_, H_ * (DN_ + DV_), KVR_);

    // rope + layout assembly
    assemble_q_kernel<<<dim3(MTOT_, H_), dim3(96), 0, stream>>>(qfull, fc, fs, qbf);
    assemble_kv_kernel<<<dim3(MTOT_, H_), dim3(128), 0, stream>>>(kvfull, kva, fc, fs, kbf, vtb);

    // flash attention
    mla_attn_kernel<<<dim3(S_ / 64, H_, B_), dim3(128), 0, stream>>>(qbf, kbf, vtb, attnb);

    // output projection straight into d_out
    gemm(attnb, wob, wo_b, out, MTOT_, DIM_, H_ * DV_);
}
